// InGramEntityLayer_64046552318127
// MI455X (gfx1250) — compile-verified, matrix operand order checked by reading, not run on hardware
//
#include <hip/hip_runtime.h>
#include <math.h>

typedef __attribute__((ext_vector_type(4)))  _Float16 v4h;
typedef __attribute__((ext_vector_type(8)))  _Float16 v8h;
typedef __attribute__((ext_vector_type(16))) _Float16 v16h;
typedef __attribute__((ext_vector_type(8)))  float    v8f;

#define DIN   32
#define DREL  32
#define DOUTD 64
#define KATT  96
#define KAGG  64
#define NH    8
#define RNUM  64

// ---------------- helpers ----------------

__device__ __forceinline__ v8f wmma(v16h a, v16h b, v8f c) {
  return __builtin_amdgcn_wmma_f32_16x16x32_f16(false, a, false, b, (short)0, c, false, false);
}

// Monotonic-bits float atomic max (works across mixed signs; init must be -inf bits)
__device__ __forceinline__ void atomicMaxF(float* a, float v) {
  int iv = __float_as_int(v);
  if (iv >= 0) atomicMax((int*)a, iv);
  else         atomicMin((unsigned int*)a, (unsigned int)iv);
}

// Build 16-half fragment from LDS f16: elements 0..7 at p[0..7], 8..15 at p[16..23]
__device__ __forceinline__ v16h fragLds(const _Float16* p) {
  v8h lo = *(const v8h*)p;
  v8h hi = *(const v8h*)(p + 16);
  v16h r;
#pragma unroll
  for (int i = 0; i < 8; ++i) { r[i] = lo[i]; r[i + 8] = hi[i]; }
  return r;
}

// Build hi/lo split fragment from global f32: K elems p[0..7] and p[16..23]
struct Frag { v16h h, l; };
__device__ __forceinline__ Frag mkfrag(const float* p) {
  float4 x0 = *(const float4*)(p + 0);
  float4 x1 = *(const float4*)(p + 4);
  float4 y0 = *(const float4*)(p + 16);
  float4 y1 = *(const float4*)(p + 20);
  float v[16] = {x0.x, x0.y, x0.z, x0.w, x1.x, x1.y, x1.z, x1.w,
                 y0.x, y0.y, y0.z, y0.w, y1.x, y1.y, y1.z, y1.w};
  Frag f;
#pragma unroll
  for (int i = 0; i < 16; ++i) {
    _Float16 hh = (_Float16)v[i];
    f.h[i] = hh;
    f.l[i] = (_Float16)(v[i] - (float)hh);
  }
  return f;
}

// ---------------- kernels ----------------

__global__ void k_init(int* counts, float* amax, float* asum, float* out, int N) {
  int i = blockIdx.x * blockDim.x + threadIdx.x;
  if (i < N * DOUTD) { counts[i] = 0; out[i] = 0.f; }
  if (i < N * NH)    { amax[i] = __int_as_float(0xFF800000); asum[i] = 0.f; }
}

__global__ void k_count(const int* __restrict__ tailI, const int* __restrict__ relI,
                        int E, int* __restrict__ counts) {
  int e = blockIdx.x * blockDim.x + threadIdx.x;
  if (e < E) atomicAdd(&counts[tailI[e] * RNUM + relI[e]], 1);
}

// self_rel[n][d] = sum_r counts[n][r]*emb_rel[r][d] / freq[n]
// one wave handles one entity (32 lanes = 32 dims): broadcast count loads
__global__ void k_selfrel(const int* __restrict__ counts, const float* __restrict__ emb_rel,
                          float* __restrict__ self_rel, int N) {
  int i = blockIdx.x * blockDim.x + threadIdx.x;
  if (i >= N * DREL) return;
  int n = i >> 5, d = i & 31;
  const int* c = counts + n * RNUM;
  float acc = 0.f, freq = 0.f;
#pragma unroll 8
  for (int r = 0; r < RNUM; ++r) {
    float cc = (float)c[r];
    freq += cc;
    acc  += cc * emb_rel[r * DREL + d];
  }
  self_rel[i] = (freq > 0.f) ? acc / freq : 0.f;
}

// Pass A: attn logits via WMMA, store logits, atomicMax into amax[tail][h]
__global__ void __launch_bounds__(256) k_attn(
    const float* __restrict__ emb_ent, const float* __restrict__ emb_rel,
    const float* __restrict__ attn_W, const float* __restrict__ attn_b,
    const float* __restrict__ attn_vec, const int* __restrict__ headI,
    const int* __restrict__ tailI, const int* __restrict__ relI,
    const float* __restrict__ self_rel, float* __restrict__ attnval,
    float* __restrict__ amax, int E, int EN)
{
  __shared__ alignas(16) _Float16 wAh[DOUTD * KATT];
  __shared__ alignas(16) _Float16 wAl[DOUTD * KATT];
  __shared__ float sb[DOUTD], sv[DOUTD];

  for (int i = threadIdx.x; i < DOUTD * KATT; i += 256) {
    float w = attn_W[i];
    _Float16 h = (_Float16)w;
    wAh[i] = h;
    wAl[i] = (_Float16)(w - (float)h);
  }
  if (threadIdx.x < DOUTD) { sb[threadIdx.x] = attn_b[threadIdx.x]; sv[threadIdx.x] = attn_vec[threadIdx.x]; }
  __syncthreads();

  int wave = threadIdx.x >> 5, lane = threadIdx.x & 31;
  int tile = blockIdx.x * 8 + wave;
  int e0 = tile * 16;
  if (e0 >= EN) return;

  int lm = lane & 15, hi = lane >> 4;
  int e = min(e0 + lm, EN - 1);
  int t, hn;
  const float* relp;
  if (e < E) { t = tailI[e]; hn = headI[e]; relp = emb_rel + (size_t)relI[e] * DREL; }
  else       { t = e - E;    hn = t;        relp = self_rel + (size_t)(e - E) * DREL; }

  // A fragments: K chunks 0..31 = ent_tail, 32..63 = ent_head, 64..95 = rel
  Frag a0 = mkfrag(emb_ent + (size_t)t  * DIN + hi * 8);
  Frag a1 = mkfrag(emb_ent + (size_t)hn * DIN + hi * 8);
  Frag a2 = mkfrag(relp + hi * 8);

#pragma unroll
  for (int nt = 0; nt < 4; ++nt) {
    int o = nt * 16 + lm;                      // output column (0..63)
    const _Float16* bhp = &wAh[o * KATT + hi * 8];
    const _Float16* blp = &wAl[o * KATT + hi * 8];
    v8f acc = {};
    {
      v16h bh = fragLds(bhp + 0),  bl = fragLds(blp + 0);
      acc = wmma(a0.h, bh, acc); acc = wmma(a0.l, bh, acc); acc = wmma(a0.h, bl, acc);
    }
    {
      v16h bh = fragLds(bhp + 32), bl = fragLds(blp + 32);
      acc = wmma(a1.h, bh, acc); acc = wmma(a1.l, bh, acc); acc = wmma(a1.h, bl, acc);
    }
    {
      v16h bh = fragLds(bhp + 64), bl = fragLds(blp + 64);
      acc = wmma(a2.h, bh, acc); acc = wmma(a2.l, bh, acc); acc = wmma(a2.h, bl, acc);
    }
    float bias = sb[o];
    float vec  = sv[o];        // attn_vec flat [h*8+d] == [o]
    int h = o >> 3;
#pragma unroll
    for (int r = 0; r < 8; ++r) {
      // D layout: element r, lanes<16 -> row r, lanes>=16 -> row r+8; col = o
      float x = acc[r] + bias;
      x = x > 0.f ? x : 0.2f * x;              // leaky_relu(0.2)
      float p = x * vec;
      p += __shfl_xor(p, 1, 32);               // reduce over d = lane bits [2:0]
      p += __shfl_xor(p, 2, 32);
      p += __shfl_xor(p, 4, 32);
      int m  = r + hi * 8;                     // edge row within tile
      int tm = __shfl(t, m, 32);               // tail of edge m (held by lane m)
      if ((lane & 7) == 0 && (e0 + m) < EN) {
        attnval[(size_t)(e0 + m) * NH + h] = p;
        atomicMaxF(&amax[(size_t)tm * NH + h], p);
      }
    }
  }
}

// Pass B: val = exp(raw - amax[tail]); accumulate attn_sum
__global__ void k_norm(float* __restrict__ attnval, const float* __restrict__ amax,
                       float* __restrict__ asum, const int* __restrict__ tailI,
                       int E, int EN) {
  int i = blockIdx.x * blockDim.x + threadIdx.x;
  if (i >= EN * NH) return;
  int e = i >> 3, h = i & 7;
  int t = (e < E) ? tailI[e] : (e - E);
  float v = expf(attnval[i] - amax[(size_t)t * NH + h]);
  attnval[i] = v;
  atomicAdd(&asum[(size_t)t * NH + h], v);
}

// Pass C: msg GEMM via WMMA, out[tail] += beta * msg
__global__ void __launch_bounds__(256) k_aggr(
    const float* __restrict__ emb_ent, const float* __restrict__ emb_rel,
    const float* __restrict__ aggr_W, const float* __restrict__ aggr_b,
    const int* __restrict__ headI, const int* __restrict__ tailI,
    const int* __restrict__ relI, const float* __restrict__ self_rel,
    const float* __restrict__ attnval, const float* __restrict__ asum,
    float* __restrict__ out, int E, int EN)
{
  __shared__ alignas(16) _Float16 wGh[DOUTD * KAGG];
  __shared__ alignas(16) _Float16 wGl[DOUTD * KAGG];
  __shared__ float sbg[DOUTD];

  for (int i = threadIdx.x; i < DOUTD * KAGG; i += 256) {
    float w = aggr_W[i];
    _Float16 h = (_Float16)w;
    wGh[i] = h;
    wGl[i] = (_Float16)(w - (float)h);
  }
  if (threadIdx.x < DOUTD) sbg[threadIdx.x] = aggr_b[threadIdx.x];
  __syncthreads();

  int wave = threadIdx.x >> 5, lane = threadIdx.x & 31;
  int tile = blockIdx.x * 8 + wave;
  int e0 = tile * 16;
  if (e0 >= EN) return;

  int lm = lane & 15, hi = lane >> 4;
  int e = min(e0 + lm, EN - 1);
  int t, hn;
  const float* relp;
  if (e < E) { t = tailI[e]; hn = headI[e]; relp = emb_rel + (size_t)relI[e] * DREL; }
  else       { t = e - E;    hn = t;        relp = self_rel + (size_t)(e - E) * DREL; }

  // cat_msg: K 0..31 = ent_head, 32..63 = rel
  Frag a0 = mkfrag(emb_ent + (size_t)hn * DIN + hi * 8);
  Frag a1 = mkfrag(relp + hi * 8);

#pragma unroll
  for (int nt = 0; nt < 4; ++nt) {
    int o = nt * 16 + lm;
    const _Float16* bhp = &wGh[o * KAGG + hi * 8];
    const _Float16* blp = &wGl[o * KAGG + hi * 8];
    v8f acc = {};
    {
      v16h bh = fragLds(bhp + 0),  bl = fragLds(blp + 0);
      acc = wmma(a0.h, bh, acc); acc = wmma(a0.l, bh, acc); acc = wmma(a0.h, bl, acc);
    }
    {
      v16h bh = fragLds(bhp + 32), bl = fragLds(blp + 32);
      acc = wmma(a1.h, bh, acc); acc = wmma(a1.l, bh, acc); acc = wmma(a1.h, bl, acc);
    }
    float bias = sbg[o];
    int h = o >> 3;
#pragma unroll
    for (int r = 0; r < 8; ++r) {
      int m  = r + hi * 8;
      int tm = __shfl(t, m, 32);
      if ((e0 + m) < EN) {
        float beta = attnval[(size_t)(e0 + m) * NH + h]
                   / (asum[(size_t)tm * NH + h] + 1e-16f);
        atomicAdd(&out[(size_t)tm * DOUTD + o], beta * (acc[r] + bias));
      }
    }
  }
}

// ---------------- launch ----------------

extern "C" void kernel_launch(void* const* d_in, const int* in_sizes, int n_in,
                              void* d_out, int out_size, void* d_ws, size_t ws_size,
                              hipStream_t stream) {
  const float* emb_ent  = (const float*)d_in[0];
  const float* emb_rel  = (const float*)d_in[1];
  const float* attn_W   = (const float*)d_in[2];
  const float* attn_b   = (const float*)d_in[3];
  const float* attn_vec = (const float*)d_in[4];
  const float* aggr_W   = (const float*)d_in[5];
  const float* aggr_b   = (const float*)d_in[6];
  const int* headI = (const int*)d_in[7];
  const int* tailI = (const int*)d_in[8];
  const int* relI  = (const int*)d_in[9];
  float* out = (float*)d_out;

  int N  = in_sizes[0] / DIN;
  int E  = in_sizes[7];
  int EN = E + N;

  char* ws = (char*)d_ws;
  size_t off = 0;
  auto take = [&](size_t bytes) -> char* {
    char* p = ws + off;
    off = (off + bytes + 255) & ~(size_t)255;
    return p;
  };
  int*   counts   = (int*)  take((size_t)N * RNUM * 4);
  float* self_rel = (float*)take((size_t)N * DREL * 4);
  float* amax     = (float*)take((size_t)N * NH * 4);
  float* asum     = (float*)take((size_t)N * NH * 4);
  float* attnval  = (float*)take((size_t)EN * NH * 4);

  int T = (EN + 15) / 16;
  int blocksT = (T + 7) / 8;

  k_init   <<<(N * DOUTD + 255) / 256, 256, 0, stream>>>(counts, amax, asum, out, N);
  k_count  <<<(E + 255) / 256,          256, 0, stream>>>(tailI, relI, E, counts);
  k_selfrel<<<(N * DREL + 255) / 256,   256, 0, stream>>>(counts, emb_rel, self_rel, N);
  k_attn   <<<blocksT,                  256, 0, stream>>>(emb_ent, emb_rel, attn_W, attn_b, attn_vec,
                                                          headI, tailI, relI, self_rel,
                                                          attnval, amax, E, EN);
  k_norm   <<<(EN * NH + 255) / 256,    256, 0, stream>>>(attnval, amax, asum, tailI, E, EN);
  k_aggr   <<<blocksT,                  256, 0, stream>>>(emb_ent, emb_rel, aggr_W, aggr_b,
                                                          headI, tailI, relI, self_rel,
                                                          attnval, asum, out, E, EN);
}
